// DELTAClassifier_75617194213680
// MI455X (gfx1250) — compile-verified, matrix operand order checked by reading, not run on hardware
//
#include <hip/hip_runtime.h>
#include <math.h>

// ---------------------------------------------------------------------------
// Problem constants (match reference)
// ---------------------------------------------------------------------------
#define NN      1024   // nodes
#define EE      4096   // edges
#define DD      256    // d_node == d_edge
#define HH      4      // heads
#define DHH     64     // head dim
#define CCOUT   16     // classes
#define NODE_KK 512
#define EDGE_KK 2048

typedef __attribute__((ext_vector_type(16))) __bf16 v16bf;
typedef __attribute__((ext_vector_type(8)))  __bf16 v8bf;
typedef __attribute__((ext_vector_type(8)))  float  v8f;

// ---------------------------------------------------------------------------
// WMMA helpers
// ---------------------------------------------------------------------------
__device__ __forceinline__ v8f wmma_bf16(v16bf a, v16bf b, v8f c) {
  // D(16x16,f32) = A(16x32,bf16) * B(32x16,bf16) + C
  return __builtin_amdgcn_wmma_f32_16x16x32_bf16(false, a, false, b,
                                                 (short)0, c, false, false);
}

// Load a 16x32 bf16 fragment in the ISA A-matrix layout from a row-major
// matrix `base` (leading dim `ld` elements), tile origin (r0, k0).
// ISA 7.12.2 (16-bit A 16x32): lanes 0-15 = rows, VGPR pairs hold
// K {0..7,16..23}; lanes 16-31 hold K {8..15,24..31}.  Same layout is used
// for B fragments by feeding B^T (N x K) through this loader.
__device__ __forceinline__ v16bf load_frag(const __bf16* __restrict__ base,
                                           int ld, int r0, int k0) {
  const int lane = threadIdx.x & 31;
  const __bf16* p = base + (size_t)(r0 + (lane & 15)) * ld
                         + (k0 + ((lane >> 4) << 3));
  v8bf lo = *(const v8bf*)(p);        // K base+0..7
  v8bf hi = *(const v8bf*)(p + 16);   // K base+16..23
  v16bf r;
#pragma unroll
  for (int i = 0; i < 8; ++i) { r[i] = lo[i]; r[i + 8] = hi[i]; }
  return r;
}

// Wave-local LDS ordering: stores and loads from one wave complete in order
// and are tracked by DScnt, so waiting DScnt==0 makes this wave's LDS writes
// visible to its own subsequent reads without a workgroup barrier.
__device__ __forceinline__ void wave_lds_fence() {
  asm volatile("s_wait_dscnt 0x0" ::: "memory");
}

// ---------------------------------------------------------------------------
// 1) Router scores: one wave per row (node rows then edge rows)
// ---------------------------------------------------------------------------
__global__ void router_scores_kernel(const float* __restrict__ nf,
                                     const float* __restrict__ ef,
                                     const float* __restrict__ Wn,
                                     const float* __restrict__ bn,
                                     const float* __restrict__ We,
                                     const float* __restrict__ be,
                                     float* __restrict__ ns,
                                     float* __restrict__ es) {
  const int wave = (int)((blockIdx.x * blockDim.x + threadIdx.x) >> 5);
  const int lane = threadIdx.x & 31;
  if (wave >= NN + EE) return;
  const float* row; const float* wv;
  if (wave < NN) { row = nf + (size_t)wave * DD;        wv = Wn; }
  else           { row = ef + (size_t)(wave - NN) * DD; wv = We; }
  float acc = 0.f;
  for (int i = lane; i < DD; i += 32) acc += row[i] * wv[i];
#pragma unroll
  for (int off = 16; off; off >>= 1) acc += __shfl_xor(acc, off, 32);
  if (lane == 0) {
    if (wave < NN) ns[wave] = acc + bn[0];
    else           es[wave - NN] = acc + be[0];
  }
}

// ---------------------------------------------------------------------------
// 2) Exact top-k by rank counting (ties by index, like lax.top_k)
// ---------------------------------------------------------------------------
__global__ void node_mask_kernel(const float* __restrict__ s, int* __restrict__ mask) {
  const int i = blockIdx.x * blockDim.x + threadIdx.x;
  if (i >= NN) return;
  const float si = s[i];
  int rank = 0;
  for (int j = 0; j < NN; ++j) {
    const float sj = s[j];
    rank += (sj > si) || (sj == si && j < i);
  }
  mask[i] = (rank < NODE_KK) ? 1 : 0;
}

__global__ void edge_mask_kernel(const float* __restrict__ s,
                                 const int* __restrict__ nmask,
                                 const int* __restrict__ src,
                                 const int* __restrict__ dst,
                                 float* __restrict__ maskf) {
  const int e = blockIdx.x * blockDim.x + threadIdx.x;
  if (e >= EE) return;
  const float se = s[e];
  int rank = 0;
  for (int j = 0; j < EE; ++j) {
    const float sj = s[j];
    rank += (sj > se) || (sj == se && j < e);
  }
  const bool keep = (rank < EDGE_KK) && nmask[src[e]] && nmask[dst[e]];
  maskf[e] = keep ? 1.0f : 0.0f;
}

// ---------------------------------------------------------------------------
// 3) xsum = nf[src] + nf[dst]  (bf16, GEMM A input)
// ---------------------------------------------------------------------------
__global__ void xsum_kernel(const float* __restrict__ nf,
                            const int* __restrict__ src,
                            const int* __restrict__ dst,
                            __bf16* __restrict__ xs) {
  const size_t idx = (size_t)blockIdx.x * blockDim.x + threadIdx.x;
  if (idx >= (size_t)EE * DD) return;
  const int e = (int)(idx >> 8);
  const int c = (int)(idx & 255);
  xs[idx] = (__bf16)(nf[(size_t)src[e] * DD + c] + nf[(size_t)dst[e] * DD + c]);
}

// ---------------------------------------------------------------------------
// 4) Weight convert+transpose: W [K x Ncol] f32 -> W^T [Ncol x K] bf16
// ---------------------------------------------------------------------------
__global__ void transw_kernel(const float* __restrict__ W, __bf16* __restrict__ WT,
                              int K, int Ncol) {
  const int idx = blockIdx.x * blockDim.x + threadIdx.x;
  if (idx >= K * Ncol) return;
  const int k = idx / Ncol, n = idx % Ncol;
  WT[(size_t)n * K + k] = (__bf16)W[idx];
}

// bf16 full transpose: V [EE x DD] -> VT [DD x EE]  (V^T per head for P@V)
__global__ void transb_kernel(const __bf16* __restrict__ V, __bf16* __restrict__ VT) {
  const size_t idx = (size_t)blockIdx.x * blockDim.x + threadIdx.x;
  if (idx >= (size_t)EE * DD) return;
  const int e = (int)(idx >> 8);
  const int c = (int)(idx & 255);
  VT[(size_t)c * EE + e] = V[idx];
}

// ---------------------------------------------------------------------------
// 5) Generic WMMA GEMM: C[MxN] = A[MxK](bf16) * B[KxN] + bias (+resid*rowscale)
//    B supplied as B^T bf16. 256 threads = 8 waves; each wave does a
//    16 x (16*NT) tile. Optional GELU, bf16 and/or f32 outputs.
// ---------------------------------------------------------------------------
template <int NT>
__global__ __launch_bounds__(256, 1)
void gemm_bf16_kernel(const __bf16* __restrict__ A,
                      const __bf16* __restrict__ BT,
                      const float* __restrict__ bias,
                      const float* __restrict__ resid,
                      const float* __restrict__ rowscale,
                      __bf16* __restrict__ Cb,
                      float* __restrict__ Cf,
                      int M, int N, int K, int act) {
  const int wave = threadIdx.x >> 5;
  const int lane = threadIdx.x & 31;
  const int wpr  = N / (16 * NT);          // waves per row-group
  const int cg   = wave % wpr;
  const int rg   = wave / wpr;
  const int rpb  = 16 * (8 / wpr);         // rows per block
  const int m0   = blockIdx.x * rpb + rg * 16;
  const int n0   = cg * 16 * NT;
  if (m0 >= M) return;

  v8f acc[NT] = {};
#pragma unroll 2
  for (int k0 = 0; k0 < K; k0 += 32) {
    v16bf a = load_frag(A, K, m0, k0);
#pragma unroll
    for (int t = 0; t < NT; ++t) {
      v16bf b = load_frag(BT, K, n0 + t * 16, k0);
      acc[t] = wmma_bf16(a, b, acc[t]);
    }
  }

  // C/D layout (ISA 7.12.2): VGPR r, lane l -> row = r + (l>=16)*8, col = l&15
  const int colL  = lane & 15;
  const int half8 = (lane >> 4) * 8;
#pragma unroll
  for (int t = 0; t < NT; ++t) {
    const int n = n0 + t * 16 + colL;
    const float bvv = bias ? bias[n] : 0.0f;
#pragma unroll
    for (int r = 0; r < 8; ++r) {
      const int m = m0 + half8 + r;
      float v = acc[t][r] + bvv;
      if (resid) v += resid[(size_t)m * N + n] * rowscale[m];
      if (act == 1) {  // tanh-approx GELU (jax.nn.gelu default)
        const float x3 = v * v * v;
        v = 0.5f * v * (1.0f + tanhf(0.7978845608028654f * (v + 0.044715f * x3)));
      }
      if (Cb) Cb[(size_t)m * N + n] = (__bf16)v;
      if (Cf) Cf[(size_t)m * N + n] = v;
    }
  }
}

// ---------------------------------------------------------------------------
// 6) Flash attention with on-the-fly adjacency masking.
//    Grid (EE/64, HH), block 128 (4 waves). Wave owns 16 query rows.
//    S tiles via WMMA, online softmax in registers (row stats replicated
//    across the 16-lane half that owns the row), P re-layout C->A via LDS
//    (wave-local, synchronized with s_wait_dscnt), P@V via WMMA against
//    pre-transposed V^T.  launch_bounds(128,1): one workgroup per SIMD pair,
//    full VGPR budget -> no spills in the hot loop.
// ---------------------------------------------------------------------------
__global__ __launch_bounds__(128, 1)
void flash_attn_kernel(const __bf16* __restrict__ qm,
                       const __bf16* __restrict__ km,
                       const __bf16* __restrict__ vt,   // [DD x EE]
                       const int* __restrict__ src,
                       const int* __restrict__ dst,
                       __bf16* __restrict__ outb) {     // [EE x DD]
  __shared__ float pbuf[4][16 * 32];
  const int wave  = threadIdx.x >> 5;
  const int lane  = threadIdx.x & 31;
  const int head  = blockIdx.y;
  const int qbase = blockIdx.x * 64 + wave * 16;
  const int half8 = (lane >> 4) * 8;
  const int col   = lane & 15;
  const int hoff  = head * DHH;

  const v16bf qa0 = load_frag(qm, DD, qbase, hoff);
  const v16bf qa1 = load_frag(qm, DD, qbase, hoff + 32);

  int qs[8], qd[8];
#pragma unroll
  for (int r = 0; r < 8; ++r) {
    const int qr = qbase + half8 + r;
    qs[r] = src[qr]; qd[r] = dst[qr];
  }

  float mrow[8], lrow[8];
#pragma unroll
  for (int r = 0; r < 8; ++r) { mrow[r] = -3.0e38f; lrow[r] = 0.0f; }
  v8f acc[4] = {};

  for (int kb = 0; kb < EE; kb += 32) {
    // ---- S = Q K^T (two 16x16 tiles, K-dim 64 = two k-steps)
    v8f s0 = {}, s1 = {};
    {
      v16bf kf;
      kf = load_frag(km, DD, kb,      hoff);      s0 = wmma_bf16(qa0, kf, s0);
      kf = load_frag(km, DD, kb,      hoff + 32); s0 = wmma_bf16(qa1, kf, s0);
      kf = load_frag(km, DD, kb + 16, hoff);      s1 = wmma_bf16(qa0, kf, s1);
      kf = load_frag(km, DD, kb + 16, hoff + 32); s1 = wmma_bf16(qa1, kf, s1);
    }

    // key endpoints for this lane's columns
    const int ks0 = src[kb + col],      kd0 = dst[kb + col];
    const int ks1 = src[kb + 16 + col], kd1 = dst[kb + 16 + col];

#pragma unroll
    for (int r = 0; r < 8; ++r) {
      const bool a0 = (qs[r] == ks0) | (qs[r] == kd0) | (qd[r] == ks0) | (qd[r] == kd0);
      const bool a1 = (qs[r] == ks1) | (qs[r] == kd1) | (qd[r] == ks1) | (qd[r] == kd1);
      const float v0 = a0 ? s0[r] * 0.125f : -1.0e9f;   // 1/sqrt(64)
      const float v1 = a1 ? s1[r] * 0.125f : -1.0e9f;
      float mt = fmaxf(v0, v1);
#pragma unroll
      for (int off = 1; off < 16; off <<= 1) mt = fmaxf(mt, __shfl_xor(mt, off, 32));
      const float mnew  = fmaxf(mrow[r], mt);
      const float scale = __expf(mrow[r] - mnew);
      const float p0 = __expf(v0 - mnew);
      const float p1 = __expf(v1 - mnew);
      float ps = p0 + p1;
#pragma unroll
      for (int off = 1; off < 16; off <<= 1) ps += __shfl_xor(ps, off, 32);
      lrow[r] = lrow[r] * scale + ps;
      mrow[r] = mnew;
#pragma unroll
      for (int t = 0; t < 4; ++t) acc[t][r] *= scale;
      s0[r] = p0; s1[r] = p1;
    }

    // ---- re-layout P: C layout -> row-major LDS -> A-fragment (bf16).
    // Producer and consumer are the same wave; LDS ops from one wave are
    // in-order, so a DScnt wait suffices (no workgroup barrier).
#pragma unroll
    for (int r = 0; r < 8; ++r) {
      pbuf[wave][(half8 + r) * 32 + col]      = s0[r];
      pbuf[wave][(half8 + r) * 32 + col + 16] = s1[r];
    }
    wave_lds_fence();
    v16bf pa;
    {
      const int arow = lane & 15;
      const int akb  = (lane >> 4) << 3;
#pragma unroll
      for (int i = 0; i < 8; ++i) {
        pa[i]     = (__bf16)pbuf[wave][arow * 32 + akb + i];
        pa[i + 8] = (__bf16)pbuf[wave][arow * 32 + akb + 16 + i];
      }
    }
    wave_lds_fence();

    // ---- O += P @ V  (B^T = V^T rows are head dims, cols are keys)
#pragma unroll
    for (int t = 0; t < 4; ++t) {
      v16bf vf = load_frag(vt, EE, hoff + t * 16, kb);
      acc[t] = wmma_bf16(pa, vf, acc[t]);
    }
  }

  // ---- normalize and store (bf16, feeds the Wo GEMM)
#pragma unroll
  for (int t = 0; t < 4; ++t) {
#pragma unroll
    for (int r = 0; r < 8; ++r) {
      const int row = qbase + half8 + r;
      const int c   = hoff + t * 16 + col;
      outb[(size_t)row * DD + c] = (__bf16)(acc[t][r] / lrow[r]);
    }
  }
}

// ---------------------------------------------------------------------------
// Host orchestration
// ---------------------------------------------------------------------------
extern "C" void kernel_launch(void* const* d_in, const int* in_sizes, int n_in,
                              void* d_out, int out_size, void* d_ws, size_t ws_size,
                              hipStream_t stream) {
  (void)in_sizes; (void)n_in; (void)out_size; (void)ws_size;

  const float* nf   = (const float*)d_in[0];
  const float* ef   = (const float*)d_in[1];
  const int*   eidx = (const int*)d_in[2];
  const int*   src  = eidx;
  const int*   dst  = eidx + EE;
  const float* Wn_r = (const float*)d_in[3];  const float* bn_r = (const float*)d_in[4];
  const float* We_r = (const float*)d_in[5];  const float* be_r = (const float*)d_in[6];
  const float* Wnp  = (const float*)d_in[7];  const float* bnp  = (const float*)d_in[8];
  const float* Wq   = (const float*)d_in[9];  const float* bq   = (const float*)d_in[10];
  const float* Wk   = (const float*)d_in[11]; const float* bk   = (const float*)d_in[12];
  const float* Wv   = (const float*)d_in[13]; const float* bv   = (const float*)d_in[14];
  const float* Wo   = (const float*)d_in[15]; const float* bo   = (const float*)d_in[16];
  const float* W1   = (const float*)d_in[17]; const float* b1   = (const float*)d_in[18];
  const float* W2   = (const float*)d_in[19]; const float* b2   = (const float*)d_in[20];
  float* out = (float*)d_out;

  // workspace carve
  char* wsp = (char*)d_ws;
  size_t off = 0;
  auto take = [&](size_t bytes) -> void* {
    size_t a = (off + 255) & ~(size_t)255;
    off = a + bytes;
    return (void*)(wsp + a);
  };
  float*  nscore = (float*)take(NN * 4);
  float*  escore = (float*)take(EE * 4);
  int*    nmask  = (int*)take(NN * 4);
  float*  maskf  = (float*)take(EE * 4);
  __bf16* xsum   = (__bf16*)take((size_t)EE * DD * 2);
  __bf16* WnpT   = (__bf16*)take((size_t)DD * DD * 2);
  __bf16* WqT    = (__bf16*)take((size_t)DD * DD * 2);
  __bf16* WkT    = (__bf16*)take((size_t)DD * DD * 2);
  __bf16* WvT    = (__bf16*)take((size_t)DD * DD * 2);
  __bf16* WoT    = (__bf16*)take((size_t)DD * DD * 2);
  __bf16* W1T    = (__bf16*)take((size_t)DD * DD * 2);
  __bf16* W2T    = (__bf16*)take((size_t)CCOUT * DD * 2);
  __bf16* hb     = (__bf16*)take((size_t)EE * DD * 2);
  __bf16* qb     = (__bf16*)take((size_t)EE * DD * 2);
  __bf16* kbuf   = (__bf16*)take((size_t)EE * DD * 2);
  __bf16* vbuf   = (__bf16*)take((size_t)EE * DD * 2);
  __bf16* vtb    = (__bf16*)take((size_t)DD * EE * 2);
  __bf16* attnb  = (__bf16*)take((size_t)EE * DD * 2);
  __bf16* o1b    = (__bf16*)take((size_t)EE * DD * 2);
  __bf16* hidb   = (__bf16*)take((size_t)EE * DD * 2);

  // 1) router scores: (NN+EE) waves
  {
    const int waves = NN + EE;
    router_scores_kernel<<<(waves * 32 + 255) / 256, 256, 0, stream>>>(
        nf, ef, Wn_r, bn_r, We_r, be_r, nscore, escore);
  }
  // 2) masks
  node_mask_kernel<<<(NN + 255) / 256, 256, 0, stream>>>(nscore, nmask);
  edge_mask_kernel<<<(EE + 255) / 256, 256, 0, stream>>>(escore, nmask, src, dst, maskf);
  // 3) xsum
  xsum_kernel<<<((size_t)EE * DD + 255) / 256, 256, 0, stream>>>(nf, src, dst, xsum);
  // 4) weights -> bf16 B^T
  transw_kernel<<<(DD * DD + 255) / 256, 256, 0, stream>>>(Wnp, WnpT, DD, DD);
  transw_kernel<<<(DD * DD + 255) / 256, 256, 0, stream>>>(Wq,  WqT,  DD, DD);
  transw_kernel<<<(DD * DD + 255) / 256, 256, 0, stream>>>(Wk,  WkT,  DD, DD);
  transw_kernel<<<(DD * DD + 255) / 256, 256, 0, stream>>>(Wv,  WvT,  DD, DD);
  transw_kernel<<<(DD * DD + 255) / 256, 256, 0, stream>>>(Wo,  WoT,  DD, DD);
  transw_kernel<<<(DD * DD + 255) / 256, 256, 0, stream>>>(W1,  W1T,  DD, DD);
  transw_kernel<<<(DD * CCOUT + 255) / 256, 256, 0, stream>>>(W2, W2T, DD, CCOUT);

  // 5) h = ef*mask + xsum@Wnp + bnp   (NT=4: 8 waves cover 32x256, grid=M/32)
  gemm_bf16_kernel<4><<<EE / 32, 256, 0, stream>>>(
      xsum, WnpT, bnp, ef, maskf, hb, (float*)nullptr, EE, DD, DD, 0);
  // q/k/v
  gemm_bf16_kernel<4><<<EE / 32, 256, 0, stream>>>(
      hb, WqT, bq, nullptr, nullptr, qb, (float*)nullptr, EE, DD, DD, 0);
  gemm_bf16_kernel<4><<<EE / 32, 256, 0, stream>>>(
      hb, WkT, bk, nullptr, nullptr, kbuf, (float*)nullptr, EE, DD, DD, 0);
  gemm_bf16_kernel<4><<<EE / 32, 256, 0, stream>>>(
      hb, WvT, bv, nullptr, nullptr, vbuf, (float*)nullptr, EE, DD, DD, 0);
  // V^T for P@V fragments
  transb_kernel<<<((size_t)EE * DD + 255) / 256, 256, 0, stream>>>(vbuf, vtb);

  // 6) flash attention
  {
    dim3 grid(EE / 64, HH);
    flash_attn_kernel<<<grid, 128, 0, stream>>>(qb, kbuf, vtb, src, dst, attnb);
  }

  // 7) output projection + MLP
  gemm_bf16_kernel<4><<<EE / 32, 256, 0, stream>>>(
      attnb, WoT, bo, nullptr, nullptr, o1b, (float*)nullptr, EE, DD, DD, 0);
  gemm_bf16_kernel<4><<<EE / 32, 256, 0, stream>>>(
      o1b, W1T, b1, nullptr, nullptr, hidb, (float*)nullptr, EE, DD, DD, 1 /*gelu*/);
  // final: N=16 -> NT=1, one wave per 16x16 tile, 8 row-groups per block
  gemm_bf16_kernel<1><<<EE / 128, 256, 0, stream>>>(
      hidb, W2T, b2, nullptr, nullptr, (__bf16*)nullptr, out, EE, CCOUT, DD, 0);
}
